// ItemEmbeddingModel_20933670601235
// MI455X (gfx1250) — matrix operand order repaired and checked by reading, not compiled, not run on hardware
//
#include <hip/hip_runtime.h>

typedef __attribute__((ext_vector_type(16))) __bf16 v16bf;
typedef __attribute__((ext_vector_type(8)))  float  v8f;

#define ED    32
#define H3    96
#define NT    6           // 96 = 6 tiles of N=16
#define TLEN  200
#define BSZ   4096
#define PVV   100001
#define WAVES 4
#define HPAD  36          // padded LDS row stride (floats), 16B aligned, bank-skewed

// ---------------------------------------------------------------------------
// Pack one tower's f32 [32,96] weight matrix into WMMA bf16 B-operand lane
// layout: tile nt covers cols n = nt*16 + (lane&15); lane half (lane>>4)
// selects K half; element i of the v16bf holds K = (lane>>4)*8 + i for i<8, 16 + (lane>>4)*8 + (i-8) for i>=8.
// ws tile layout: [mat*6+nt][lane][i], mat = tower*2 + {Wx=0, Wh=1}.
// ---------------------------------------------------------------------------
__global__ void pack_weights(const float* __restrict__ iWx, const float* __restrict__ iWh,
                             const float* __restrict__ pWx, const float* __restrict__ pWh,
                             const float* __restrict__ dWx, const float* __restrict__ dWh,
                             __bf16* __restrict__ ws) {
  const int mat = blockIdx.x / NT;   // 0..5
  const int nt  = blockIdx.x % NT;   // 0..5
  const float* W = (mat == 0) ? iWx : (mat == 1) ? iWh :
                   (mat == 2) ? pWx : (mat == 3) ? pWh :
                   (mat == 4) ? dWx : dWh;
  const int lane = threadIdx.x >> 4;   // 0..31
  const int i    = threadIdx.x & 15;   // 0..15
  const int n = nt * 16 + (lane & 15);
  const int k = (i < 8) ? (lane >> 4) * 8 + i : 16 + (lane >> 4) * 8 + (i - 8);
  ws[blockIdx.x * 512 + lane * 16 + i] = (__bf16)W[k * H3 + n];
}

__device__ inline v16bf pack16(float4 a, float4 b, float4 c, float4 d) {
  v16bf v;
  v[0]  = (__bf16)a.x; v[1]  = (__bf16)a.y; v[2]  = (__bf16)a.z; v[3]  = (__bf16)a.w;
  v[4]  = (__bf16)b.x; v[5]  = (__bf16)b.y; v[6]  = (__bf16)b.z; v[7]  = (__bf16)b.w;
  v[8]  = (__bf16)c.x; v[9]  = (__bf16)c.y; v[10] = (__bf16)c.z; v[11] = (__bf16)c.w;
  v[12] = (__bf16)d.x; v[13] = (__bf16)d.y; v[14] = (__bf16)d.z; v[15] = (__bf16)d.w;
  return v;
}

__device__ inline v8f splat8(float v) { v8f r = {v, v, v, v, v, v, v, v}; return r; }

// Single-instruction v_rcp_f32 (~1 ulp): avoids the IEEE div_scale/div_fixup
// chain the compiler emits for '/'. Gate nonlinearities tolerate this easily.
__device__ inline float fast_rcp(float x) { return __builtin_amdgcn_rcpf(x); }
__device__ inline float sigm(float x) { return fast_rcp(1.0f + __expf(-x)); }
__device__ inline float tanh_fast(float x) {
  x = fminf(15.0f, fmaxf(-15.0f, x));
  float e = __expf(2.0f * x);
  return (e - 1.0f) * fast_rcp(e + 1.0f);
}

// ---------------------------------------------------------------------------
// One fused GRU tower. Each wave owns 16 batch rows (WMMA M=16) and a private
// LDS region for h [16][32] f32; it loops over all T steps with no barriers
// (same-wave LDS ops are in-order on CDNA5). 12 v_wmma_f32_16x16x32_bf16 per
// step: z/r gate tiles accumulate both x@Wx and h@Wh into one accumulator;
// the candidate tiles keep xh and rh separate (needed for r*rh).
// MODE: 0 = item (no mask), 1 = price (normalize->index, mask pn!=0),
//       2 = dept (mask id!=0).
// ---------------------------------------------------------------------------
template <int MODE>
__global__ __launch_bounds__(WAVES * 32)
void gru_tower(const int*   __restrict__ ids,
               const float* __restrict__ prices,
               const float* __restrict__ table,
               const __bf16* __restrict__ wpack,   // this tower: 6 Wx tiles, 6 Wh tiles
               const float* __restrict__ bias,     // [2,96]
               const float* __restrict__ pmean,
               const float* __restrict__ pvar,
               float* __restrict__ out, int colOff)
{
  __shared__ float hbuf[WAVES][16][HPAD];
  __shared__ float mbuf[WAVES][16];

  const int wave = threadIdx.x >> 5;
  const int lane = threadIdx.x & 31;
  const int r16  = lane & 15;       // row (A) / column (B,C,D) within tile
  const int kh   = lane >> 4;       // K-half for A/B, row-half for C/D
  const int rowBase = (blockIdx.x * WAVES + wave) * 16;
  const int myrow   = rowBase + r16;

  // Hold all 12 weight tiles in VGPRs for the whole scan.
  v16bf wx[NT], wh[NT];
#pragma unroll
  for (int nt = 0; nt < NT; ++nt) {
    wx[nt] = *(const v16bf*)(wpack + (size_t)nt * 512 + lane * 16);
    wh[nt] = *(const v16bf*)(wpack + (size_t)(NT + nt) * 512 + lane * 16);
  }

  // Bias per lane column (C/D layout: col = r16 within tile), folded into acc init.
  float bzr[4];
#pragma unroll
  for (int nt = 0; nt < 4; ++nt) bzr[nt] = bias[nt * 16 + r16] + bias[H3 + nt * 16 + r16];
  const float bx4 = bias[64 + r16],      bx5 = bias[80 + r16];
  const float bh4 = bias[H3 + 64 + r16], bh5 = bias[H3 + 80 + r16];

  float mean = 0.f, rstd = 1.f;
  if (MODE == 1) { mean = pmean[0]; rstd = rsqrtf(pvar[0]); }

  // h0 = 0
  for (int i = lane; i < 16 * HPAD; i += 32) (&hbuf[wave][0][0])[i] = 0.0f;

  for (int t = 0; t < TLEN; ++t) {
    // ---- per-row gather index + step mask ---------------------------------
    int g; float mask;
    if (MODE == 0) {
      g = ids[(size_t)myrow * TLEN + t]; mask = 1.0f;
    } else if (MODE == 1) {
      float pn = (prices[(size_t)myrow * TLEN + t] - mean) * rstd;
      mask = (pn != 0.0f) ? 1.0f : 0.0f;
      int gi = (int)pn;                    // trunc toward zero, like astype(int32)
      gi = gi < 0 ? 0 : gi; gi = gi > (PVV - 1) ? (PVV - 1) : gi;
      g = gi;
    } else {
      g = ids[(size_t)myrow * TLEN + t]; mask = (g != 0) ? 1.0f : 0.0f;
    }
    if (lane < 16) mbuf[wave][r16] = mask;   // same-wave DS, in-order

    // ---- A_x: gather embedding row in WMMA A layout -----------------------
    const float4* e = (const float4*)(table + (size_t)g * ED);
    v16bf ax = pack16(e[kh * 2 + 0], e[kh * 2 + 1], e[kh * 2 + 4], e[kh * 2 + 5]);

    // ---- A_h: read h from LDS in WMMA A layout ----------------------------
    const float4* hp = (const float4*)&hbuf[wave][r16][0];
    v16bf ah = pack16(hp[kh * 2 + 0], hp[kh * 2 + 1], hp[kh * 2 + 4], hp[kh * 2 + 5]);

    // ---- projections: 12 WMMAs -------------------------------------------
    v8f accZR[4], accXH[2], accRH[2];
#pragma unroll
    for (int nt = 0; nt < 4; ++nt) accZR[nt] = splat8(bzr[nt]);
    accXH[0] = splat8(bx4); accXH[1] = splat8(bx5);
    accRH[0] = splat8(bh4); accRH[1] = splat8(bh5);

#pragma unroll
    for (int nt = 0; nt < 4; ++nt) {
      accZR[nt] = __builtin_amdgcn_wmma_f32_16x16x32_bf16(
          false, ax, false, wx[nt], (short)0, accZR[nt], false, false);
      accZR[nt] = __builtin_amdgcn_wmma_f32_16x16x32_bf16(
          false, ah, false, wh[nt], (short)0, accZR[nt], false, false);
    }
#pragma unroll
    for (int c = 0; c < 2; ++c) {
      accXH[c] = __builtin_amdgcn_wmma_f32_16x16x32_bf16(
          false, ax, false, wx[4 + c], (short)0, accXH[c], false, false);
      accRH[c] = __builtin_amdgcn_wmma_f32_16x16x32_bf16(
          false, ah, false, wh[4 + c], (short)0, accRH[c], false, false);
    }

    // ---- gates + h update (C/D layout: this lane owns rows 8*kh..+7,
    //      hidden units j = r16 and r16+16) ---------------------------------
#pragma unroll
    for (int c = 0; c < 2; ++c) {
      const int j = r16 + 16 * c;
#pragma unroll
      for (int r = 0; r < 8; ++r) {
        const int m = r + 8 * kh;
        float hprev = hbuf[wave][m][j];
        float z  = sigm(accZR[c][r]);
        float rg = sigm(accZR[2 + c][r]);
        float cand = tanh_fast(accXH[c][r] + rg * accRH[c][r]);
        float hn = z * hprev + (1.0f - z) * cand;
        float mk = mbuf[wave][m];
        hbuf[wave][m][j] = (mk != 0.0f) ? hn : hprev;
      }
    }
    // keep the compiler from sinking/hoisting LDS ops across the step edge;
    // HW keeps same-wave DS in order, the wait is belt-and-braces.
    asm volatile("s_wait_dscnt 0x0" ::: "memory");
  }

  // ---- write final hidden state to out[:, colOff:colOff+32] ---------------
  for (int i = lane; i < 16 * ED; i += 32) {
    const int m = i >> 5, j = i & 31;
    out[(size_t)(rowBase + m) * H3 + colOff + j] = hbuf[wave][m][j];
  }
}

// ---------------------------------------------------------------------------
extern "C" void kernel_launch(void* const* d_in, const int* in_sizes, int n_in,
                              void* d_out, int out_size, void* d_ws, size_t ws_size,
                              hipStream_t stream) {
  const int*   item_id   = (const int*)  d_in[0];
  const float* price     = (const float*)d_in[1];
  const int*   dept_id   = (const int*)  d_in[2];
  const float* item_tab  = (const float*)d_in[3];
  const float* price_tab = (const float*)d_in[4];
  const float* dept_tab  = (const float*)d_in[5];
  const float* iWx = (const float*)d_in[6];
  const float* iWh = (const float*)d_in[7];
  const float* ib  = (const float*)d_in[8];
  const float* pWx = (const float*)d_in[9];
  const float* pWh = (const float*)d_in[10];
  const float* pb  = (const float*)d_in[11];
  const float* dWx = (const float*)d_in[12];
  const float* dWh = (const float*)d_in[13];
  const float* db  = (const float*)d_in[14];
  const float* pmean = (const float*)d_in[15];
  const float* pvar  = (const float*)d_in[16];
  float*  out   = (float*)d_out;
  __bf16* wpack = (__bf16*)d_ws;           // 36 tiles * 512 bf16 = 36 KB

  pack_weights<<<36, 512, 0, stream>>>(iWx, iWh, pWx, pWh, dWx, dWh, wpack);

  dim3 grid(BSZ / (WAVES * 16));
  dim3 blk(WAVES * 32);
  gru_tower<0><<<grid, blk, 0, stream>>>(item_id, nullptr, item_tab,
                                         wpack + 0 * 12 * 512, ib, pmean, pvar, out, 0);
  gru_tower<1><<<grid, blk, 0, stream>>>(nullptr, price, price_tab,
                                         wpack + 1 * 12 * 512, pb, pmean, pvar, out, 32);
  gru_tower<2><<<grid, blk, 0, stream>>>(dept_id, nullptr, dept_tab,
                                         wpack + 2 * 12 * 512, db, pmean, pvar, out, 64);
}
